// EfficientAttentionScore_46986942218813
// MI455X (gfx1250) — matrix operand order repaired and checked by reading, not compiled, not run on hardware
//
#include <hip/hip_runtime.h>
#include <hip/hip_bf16.h>

#define B_   8
#define C_   256
#define N_   16384
#define CK_  256
#define CV_  256

typedef __attribute__((ext_vector_type(16))) __bf16 v16bf;
typedef __attribute__((ext_vector_type(8)))  float  v8f;
typedef unsigned int u32x4 __attribute__((ext_vector_type(4)));
typedef int          i32x4 __attribute__((ext_vector_type(4)));
typedef int          i32x8 __attribute__((ext_vector_type(8)));

union FragU { v16bf v; unsigned int u[8]; unsigned short h[16]; };

// ---- LDS tile geometry (leading dims chosen for TDM pad fields + bank spread)
#define WLD 264   // W tile rows: 256 data + 8 pad elems  (132 dw stride, =4 mod 64)
#define XLD 132   // x tile rows: 128 data + 4 pad elems  (66 dw stride,  =2 mod 64)
#define PLD 132   // p tile rows (VALU-written)
#define VLD 136   // v tile rows: 128 data + 8 pad elems
#define SLD 132   // f32 score tile

// ---- LDS byte offsets (dynamic smem assumed at LDS offset 0: no static __shared__)
#define OFF_WT   0
#define SZ_WT    (128 * WLD * 2)            // 67584
#define OFF_X0   SZ_WT
#define SZ_X     (32 * XLD * 2)             // 8448
#define OFF_X1   (OFF_X0 + SZ_X)
#define OFF_XEND (OFF_X1 + SZ_X)            // 84480
// keys-stats kernel:
#define OFF_ST   OFF_XEND                   // 128*132*4 = 67584
#define OFF_BK   (OFF_ST + 128 * SLD * 4)   // 152064
#define OFF_RM   (OFF_BK + 512)             // 152576
#define OFF_RL2  (OFF_RM + 1024)            // 153600
#define SMEM_S   (OFF_RL2 + 1024)           // 154624
// values kernel:
#define SMEM_V   (OFF_XEND + 512)           // 84992
// context kernel:
#define OFF_PB   OFF_XEND                   // 128*132*2 = 33792
#define OFF_VB   (OFF_PB + 128 * PLD * 2)   // 118272; 128*136*2 = 34816
#define OFF_MB   (OFF_VB + 128 * VLD * 2)   // 153088
#define OFF_RLS  (OFF_MB + 512)             // 153600
#define SMEM_C   (OFF_RLS + 512)            // 154112

__device__ __forceinline__ unsigned short f32_to_bf16(float f) {
    union { float f; unsigned int u; } c; c.f = f;
    unsigned int r = c.u + 0x7FFFu + ((c.u >> 16) & 1u);   // round-to-nearest-even
    return (unsigned short)(r >> 16);
}

// Tensor Data Mover: 2-D tile of 2-byte elements, global -> LDS, with hardware
// LDS row padding. Issued by one wave; completion tracked on TENSORcnt.
__device__ __forceinline__ void tdm_load_2d(unsigned lds_off, const void* gptr,
                                            unsigned tensor_d0, unsigned tensor_d1,
                                            unsigned tile_d0, unsigned tile_d1,
                                            unsigned stride0_elems,
                                            unsigned pad_interval, unsigned pad_amount) {
    unsigned long long ga = (unsigned long long)gptr;
    u32x4 g0;
    g0[0] = 1u;                                            // count=1, user descriptor
    g0[1] = lds_off;                                       // lds_addr (bytes)
    g0[2] = (unsigned)ga;                                  // global_addr[31:0]
    g0[3] = (unsigned)((ga >> 32) & 0x01FFFFFFull) | (2u << 30);  // addr[56:32], type=2
    i32x8 g1;
    g1[0] = (int)((1u << 16)                               // data_size = 2 bytes
                  | (1u << 20)                             // pad_enable
                  | (pad_interval << 22) | (pad_amount << 25));
    g1[1] = (int)((tensor_d0 & 0xFFFFu) << 16);            // tensor_dim0[15:0]
    g1[2] = (int)((tensor_d0 >> 16) | ((tensor_d1 & 0xFFFFu) << 16));
    g1[3] = (int)((tensor_d1 >> 16) | ((tile_d0 & 0xFFFFu) << 16));
    g1[4] = (int)(tile_d1 & 0xFFFFu);                      // tile_dim1, tile_dim2=0
    g1[5] = (int)stride0_elems;                            // tensor_dim0_stride[31:0]
    g1[6] = 0;
    g1[7] = 0;
    i32x4 z4 = {0, 0, 0, 0};
#if __clang_major__ >= 23
    i32x8 z8 = {0, 0, 0, 0, 0, 0, 0, 0};
    __builtin_amdgcn_tensor_load_to_lds(g0, g1, z4, z4, z8, 0);
#else
    __builtin_amdgcn_tensor_load_to_lds(g0, g1, z4, z4, 0);
#endif
}

// A fragment: 16(M) x 32(K) bf16; LDS row-major over M, ld elems (even).
__device__ __forceinline__ v16bf load_frag_a(const unsigned short* sh, int ld,
                                             int mbase, int kbase, int lane) {
    FragU f;
    const int m  = mbase + (lane & 15);
    const int k0 = kbase + ((lane >> 4) << 4);
    const unsigned int* p = (const unsigned int*)(sh + m * ld + k0);
#pragma unroll
    for (int i = 0; i < 8; ++i) f.u[i] = p[i];
    return f.v;
}

// B fragment: 32(K) x 16(N) bf16; LDS row-major over K.
__device__ __forceinline__ v16bf load_frag_b_rm(const unsigned short* sh, int ld,
                                                int kbase, int nbase, int lane) {
    FragU f;
    const int n  = nbase + (lane & 15);
    const int k0 = kbase + ((lane >> 4) << 4);
#pragma unroll
    for (int i = 0; i < 16; ++i) f.h[i] = sh[(k0 + i) * ld + n];
    return f.v;
}

// B fragment where the LDS tile is transposed: B[k][n] = sh[n*ld + k].
__device__ __forceinline__ v16bf load_frag_b_cm(const unsigned short* sh, int ld,
                                                int kbase, int nbase, int lane) {
    FragU f;
    const int n  = nbase + (lane & 15);
    const int k0 = kbase + ((lane >> 4) << 4);
    const unsigned int* p = (const unsigned int*)(sh + n * ld + k0);
#pragma unroll
    for (int i = 0; i < 8; ++i) f.u[i] = p[i];
    return f.v;
}

__device__ __forceinline__ v8f wmma_bf16(v16bf a, v16bf b, v8f c) {
    return __builtin_amdgcn_wmma_f32_16x16x32_bf16(false, a, false, b, (short)0, c,
                                                   false, false);
}

// ---------------------------------------------------------------- converts ---
__global__ void k_cvt_w(const float* __restrict__ Wk, const float* __restrict__ Wv,
                        unsigned short* __restrict__ Wkb, unsigned short* __restrict__ Wvb) {
    int i = blockIdx.x * 256 + threadIdx.x;           // 65536 total
    Wkb[i] = f32_to_bf16(Wk[i]);
    Wvb[i] = f32_to_bf16(Wv[i]);
}

__global__ void k_cvt_x(const float* __restrict__ x, unsigned short* __restrict__ xb) {
    size_t i = (size_t)(blockIdx.x * 256 + threadIdx.x) * 4;   // 33,554,432 elems / 4
    float4 f = *(const float4*)(x + i);
    unsigned int lo = (unsigned int)f32_to_bf16(f.x) | ((unsigned int)f32_to_bf16(f.y) << 16);
    unsigned int hi = (unsigned int)f32_to_bf16(f.z) | ((unsigned int)f32_to_bf16(f.w) << 16);
    uint2 o; o.x = lo; o.y = hi;
    *(uint2*)(xb + i) = o;
}

// -------------------------------------------------------- keys stats (pass 1) ---
// grid (N/128, 2, B): s = Wk[m0:+128]*x + bk over a 128x128 tile; emit per-row
// chunk max & sum(exp(s-max)) -> mpart/lpart[b][k][chunk]. TDM-fed, double-buffered.
__global__ __launch_bounds__(256) void k_keys_stats(
    const unsigned short* __restrict__ Wkb, const unsigned short* __restrict__ xb,
    const float* __restrict__ bk, float* __restrict__ mpart, float* __restrict__ lpart) {
    extern __shared__ char smem[];
    unsigned short* wtile = (unsigned short*)(smem + OFF_WT);
    unsigned short* xt0   = (unsigned short*)(smem + OFF_X0);
    unsigned short* xt1   = (unsigned short*)(smem + OFF_X1);
    float* stile = (float*)(smem + OFF_ST);
    float* bkS   = (float*)(smem + OFF_BK);
    float* redm  = (float*)(smem + OFF_RM);
    float* redl  = (float*)(smem + OFF_RL2);

    const int tid = threadIdx.x, lane = tid & 31, w = tid >> 5;
    const int wm = w & 1, wn = w >> 1;
    const int b = blockIdx.z, m0 = blockIdx.y * 128, n0 = blockIdx.x * 128;
    const unsigned short* xbp = xb + (size_t)b * C_ * N_;

    if (tid < 32) {   // wave 0 drives the Tensor Data Mover
        tdm_load_2d(OFF_WT, Wkb + (size_t)m0 * C_, C_, 128, C_, 128, C_, 6, 3);
        tdm_load_2d(OFF_X0, xbp + n0, N_, C_, 128, 32, N_, 5, 1);
    }
    if (tid < 128) bkS[tid] = bk[m0 + tid];

    v8f z = {0.f, 0.f, 0.f, 0.f, 0.f, 0.f, 0.f, 0.f};
    v8f acc[4][2];
#pragma unroll
    for (int mi = 0; mi < 4; ++mi)
#pragma unroll
        for (int nj = 0; nj < 2; ++nj) acc[mi][nj] = z;

    for (int kk = 0; kk < C_; kk += 32) {
        if (tid < 32) {
            if (kk + 32 < C_) {
                unsigned dst = (((kk >> 5) + 1) & 1) ? OFF_X1 : OFF_X0;
                tdm_load_2d(dst, xbp + (size_t)(kk + 32) * N_ + n0, N_, C_, 128, 32, N_, 5, 1);
                __builtin_amdgcn_s_wait_tensorcnt(1);   // all but in-flight tile done
            } else {
                __builtin_amdgcn_s_wait_tensorcnt(0);
            }
        }
        __syncthreads();
        const unsigned short* xt = ((kk >> 5) & 1) ? xt1 : xt0;
        v16bf a[4], bb[2];
#pragma unroll
        for (int mi = 0; mi < 4; ++mi) a[mi] = load_frag_a(wtile, WLD, wm * 64 + mi * 16, kk, lane);
#pragma unroll
        for (int nj = 0; nj < 2; ++nj) bb[nj] = load_frag_b_rm(xt, XLD, 0, wn * 32 + nj * 16, lane);
#pragma unroll
        for (int mi = 0; mi < 4; ++mi)
#pragma unroll
            for (int nj = 0; nj < 2; ++nj) acc[mi][nj] = wmma_bf16(a[mi], bb[nj], acc[mi][nj]);
        __syncthreads();
    }

    const int hi8 = (lane >> 4) << 3, col = lane & 15;
#pragma unroll
    for (int mi = 0; mi < 4; ++mi)
#pragma unroll
        for (int nj = 0; nj < 2; ++nj)
#pragma unroll
            for (int i = 0; i < 8; ++i) {
                int r = wm * 64 + mi * 16 + i + hi8;
                int c = wn * 32 + nj * 16 + col;
                stile[r * SLD + c] = acc[mi][nj][i] + bkS[r];
            }
    __syncthreads();
    {
        int r = tid & 127, half = tid >> 7;
        const float* row = stile + r * SLD + half * 64;
        float mx = -3.0e38f;
        for (int c2 = 0; c2 < 64; ++c2) mx = fmaxf(mx, row[c2]);
        float s = 0.f;
        for (int c2 = 0; c2 < 64; ++c2) s += __expf(row[c2] - mx);
        redm[half * 128 + r] = mx;
        redl[half * 128 + r] = s;
    }
    __syncthreads();
    if (tid < 128) {
        float m1 = redm[tid], m2 = redm[128 + tid];
        float mg = fmaxf(m1, m2);
        float lg = redl[tid] * __expf(m1 - mg) + redl[128 + tid] * __expf(m2 - mg);
        size_t o = ((size_t)(b * CK_ + m0 + tid)) * (N_ / 128) + blockIdx.x;
        mpart[o] = mg; lpart[o] = lg;
    }
}

// ------------------------------------------------------------ values GEMM ---
__global__ __launch_bounds__(256) void k_values(
    const unsigned short* __restrict__ Wvb, const unsigned short* __restrict__ xb,
    const float* __restrict__ bv, unsigned short* __restrict__ vb) {
    extern __shared__ char smem[];
    unsigned short* wtile = (unsigned short*)(smem + OFF_WT);
    unsigned short* xt0   = (unsigned short*)(smem + OFF_X0);
    unsigned short* xt1   = (unsigned short*)(smem + OFF_X1);
    float* bvS = (float*)(smem + OFF_XEND);

    const int tid = threadIdx.x, lane = tid & 31, w = tid >> 5;
    const int wm = w & 1, wn = w >> 1;
    const int b = blockIdx.z, m0 = blockIdx.y * 128, n0 = blockIdx.x * 128;
    const unsigned short* xbp = xb + (size_t)b * C_ * N_;

    if (tid < 32) {
        tdm_load_2d(OFF_WT, Wvb + (size_t)m0 * C_, C_, 128, C_, 128, C_, 6, 3);
        tdm_load_2d(OFF_X0, xbp + n0, N_, C_, 128, 32, N_, 5, 1);
    }
    if (tid < 128) bvS[tid] = bv[m0 + tid];

    v8f z = {0.f, 0.f, 0.f, 0.f, 0.f, 0.f, 0.f, 0.f};
    v8f acc[4][2];
#pragma unroll
    for (int mi = 0; mi < 4; ++mi)
#pragma unroll
        for (int nj = 0; nj < 2; ++nj) acc[mi][nj] = z;

    for (int kk = 0; kk < C_; kk += 32) {
        if (tid < 32) {
            if (kk + 32 < C_) {
                unsigned dst = (((kk >> 5) + 1) & 1) ? OFF_X1 : OFF_X0;
                tdm_load_2d(dst, xbp + (size_t)(kk + 32) * N_ + n0, N_, C_, 128, 32, N_, 5, 1);
                __builtin_amdgcn_s_wait_tensorcnt(1);
            } else {
                __builtin_amdgcn_s_wait_tensorcnt(0);
            }
        }
        __syncthreads();
        const unsigned short* xt = ((kk >> 5) & 1) ? xt1 : xt0;
        v16bf a[4], bb[2];
#pragma unroll
        for (int mi = 0; mi < 4; ++mi) a[mi] = load_frag_a(wtile, WLD, wm * 64 + mi * 16, kk, lane);
#pragma unroll
        for (int nj = 0; nj < 2; ++nj) bb[nj] = load_frag_b_rm(xt, XLD, 0, wn * 32 + nj * 16, lane);
#pragma unroll
        for (int mi = 0; mi < 4; ++mi)
#pragma unroll
            for (int nj = 0; nj < 2; ++nj) acc[mi][nj] = wmma_bf16(a[mi], bb[nj], acc[mi][nj]);
        __syncthreads();
    }

    const int hi8 = (lane >> 4) << 3, col = lane & 15;
#pragma unroll
    for (int mi = 0; mi < 4; ++mi)
#pragma unroll
        for (int nj = 0; nj < 2; ++nj)
#pragma unroll
            for (int i = 0; i < 8; ++i) {
                int rl = wm * 64 + mi * 16 + i + hi8;
                int c  = n0 + wn * 32 + nj * 16 + col;
                float v = acc[mi][nj][i] + bvS[rl];
                vb[((size_t)b * CV_ + m0 + rl) * N_ + c] = f32_to_bf16(v);
            }
}

// ------------------------------------------------------- stats reduce ---
__global__ void k_reduce_stats(const float* __restrict__ mpart, const float* __restrict__ lpart,
                               float* __restrict__ mfull, float* __restrict__ lrec) {
    int t = blockIdx.x * 256 + threadIdx.x;            // 2048 = B*CK
    const int CH = N_ / 128;
    const float* mp = mpart + (size_t)t * CH;
    const float* lp = lpart + (size_t)t * CH;
    float m = -3.0e38f, l = 0.f;
    for (int c = 0; c < CH; ++c) {
        float mc = mp[c], lc = lp[c];
        float mn = fmaxf(m, mc);
        l = l * __expf(m - mn) + lc * __expf(mc - mn);
        m = mn;
    }
    mfull[t] = m;
    lrec[t]  = 1.0f / l;
}

__global__ void k_zero(float* __restrict__ out) {
    out[(size_t)blockIdx.x * 256 + threadIdx.x] = 0.f;
}

// ----------------------------------------------------- fused context (pass 2) ---
// grid (N/1024, 4, B). blockIdx.y: bit0 -> k half, bit1 -> v half.
// Per 128-col sub-chunk: GEMM1 s=Wk*x, p=exp(s-m)/l -> LDS bf16; GEMM2 ctx += p*v^T.
// All global->LDS movement via TDM; f32 atomics combine across N chunks.
__global__ __launch_bounds__(256) void k_context(
    const unsigned short* __restrict__ Wkb, const unsigned short* __restrict__ xb,
    const unsigned short* __restrict__ vb, const float* __restrict__ bk,
    const float* __restrict__ mfull, const float* __restrict__ lrec,
    float* __restrict__ out) {
    extern __shared__ char smem[];
    unsigned short* wtile = (unsigned short*)(smem + OFF_WT);
    unsigned short* xt0   = (unsigned short*)(smem + OFF_X0);
    unsigned short* xt1   = (unsigned short*)(smem + OFF_X1);
    unsigned short* pbuf  = (unsigned short*)(smem + OFF_PB);
    unsigned short* vbuf  = (unsigned short*)(smem + OFF_VB);
    float* mbS = (float*)(smem + OFF_MB);
    float* rlS = (float*)(smem + OFF_RLS);

    const int tid = threadIdx.x, lane = tid & 31, w = tid >> 5;
    const int wm = w & 1, wn = w >> 1;
    const int b = blockIdx.z;
    const int m0 = (blockIdx.y & 1) * 128;
    const int v0 = (blockIdx.y >> 1) * 128;
    const int nbase0 = blockIdx.x * 1024;
    const unsigned short* xbp = xb + (size_t)b * C_ * N_;
    const unsigned short* vbp = vb + (size_t)b * CV_ * N_;

    if (tid < 32)
        tdm_load_2d(OFF_WT, Wkb + (size_t)m0 * C_, C_, 128, C_, 128, C_, 6, 3);
    if (tid < 128) {
        int r = m0 + tid;
        mbS[tid] = mfull[b * CK_ + r] - bk[r];
        rlS[tid] = lrec[b * CK_ + r];
    }

    v8f z = {0.f, 0.f, 0.f, 0.f, 0.f, 0.f, 0.f, 0.f};
    v8f ctx[4][2];
#pragma unroll
    for (int mi = 0; mi < 4; ++mi)
#pragma unroll
        for (int vj = 0; vj < 2; ++vj) ctx[mi][vj] = z;

    const int hi8 = (lane >> 4) << 3, col = lane & 15;

    for (int sc = 0; sc < 8; ++sc) {
        const int n0 = nbase0 + sc * 128;
        if (tid < 32) {   // v tile + first x tile; overlap with compute below
            tdm_load_2d(OFF_VB, vbp + (size_t)v0 * N_ + n0, N_, CV_, 128, 128, N_, 5, 3);
            tdm_load_2d(OFF_X0, xbp + n0, N_, C_, 128, 32, N_, 5, 1);
        }
        v8f pacc[4][2];
#pragma unroll
        for (int mi = 0; mi < 4; ++mi)
#pragma unroll
            for (int nj = 0; nj < 2; ++nj) pacc[mi][nj] = z;

        for (int kk = 0; kk < C_; kk += 32) {                 // GEMM1: scores
            if (tid < 32) {
                if (kk + 32 < C_) {
                    unsigned dst = (((kk >> 5) + 1) & 1) ? OFF_X1 : OFF_X0;
                    tdm_load_2d(dst, xbp + (size_t)(kk + 32) * N_ + n0, N_, C_, 128, 32, N_, 5, 1);
                    __builtin_amdgcn_s_wait_tensorcnt(1);
                } else {
                    __builtin_amdgcn_s_wait_tensorcnt(0);
                }
            }
            __syncthreads();
            const unsigned short* xt = ((kk >> 5) & 1) ? xt1 : xt0;
            v16bf a[4], bb[2];
#pragma unroll
            for (int mi = 0; mi < 4; ++mi) a[mi] = load_frag_a(wtile, WLD, wm * 64 + mi * 16, kk, lane);
#pragma unroll
            for (int nj = 0; nj < 2; ++nj) bb[nj] = load_frag_b_rm(xt, XLD, 0, wn * 32 + nj * 16, lane);
#pragma unroll
            for (int mi = 0; mi < 4; ++mi)
#pragma unroll
                for (int nj = 0; nj < 2; ++nj) pacc[mi][nj] = wmma_bf16(a[mi], bb[nj], pacc[mi][nj]);
            __syncthreads();
        }

        // p = exp(s + bk - m) / l  ->  pbuf (bf16 A operand of GEMM2)
#pragma unroll
        for (int mi = 0; mi < 4; ++mi)
#pragma unroll
            for (int nj = 0; nj < 2; ++nj)
#pragma unroll
                for (int i = 0; i < 8; ++i) {
                    int r = wm * 64 + mi * 16 + i + hi8;
                    int c = wn * 32 + nj * 16 + col;
                    float p = __expf(pacc[mi][nj][i] - mbS[r]) * rlS[r];
                    pbuf[r * PLD + c] = f32_to_bf16(p);
                }
        __syncthreads();

        for (int kn = 0; kn < 128; kn += 32) {                // GEMM2: p * v^T
            v16bf a2[4], b2[2];
#pragma unroll
            for (int mi = 0; mi < 4; ++mi) a2[mi] = load_frag_a(pbuf, PLD, wm * 64 + mi * 16, kn, lane);
#pragma unroll
            for (int vj = 0; vj < 2; ++vj) b2[vj] = load_frag_b_cm(vbuf, VLD, kn, wn * 32 + vj * 16, lane);
#pragma unroll
            for (int mi = 0; mi < 4; ++mi)
#pragma unroll
                for (int vj = 0; vj < 2; ++vj) ctx[mi][vj] = wmma_bf16(a2[mi], b2[vj], ctx[mi][vj]);
        }
        __syncthreads();
    }

#pragma unroll
    for (int mi = 0; mi < 4; ++mi)
#pragma unroll
        for (int vj = 0; vj < 2; ++vj)
#pragma unroll
            for (int i = 0; i < 8; ++i) {
                int r = m0 + wm * 64 + mi * 16 + i + hi8;
                int c = v0 + wn * 32 + vj * 16 + col;
                atomicAdd(out + ((size_t)b * CK_ + r) * CV_ + c, ctx[mi][vj][i]);
            }
}

// ------------------------------------------------------------------- launch ---
extern "C" void kernel_launch(void* const* d_in, const int* in_sizes, int n_in,
                              void* d_out, int out_size, void* d_ws, size_t ws_size,
                              hipStream_t stream) {
    (void)in_sizes; (void)n_in; (void)out_size; (void)ws_size;
    const float* x  = (const float*)d_in[0];
    const float* Wk = (const float*)d_in[1];
    const float* bk = (const float*)d_in[2];
    const float* Wv = (const float*)d_in[3];
    const float* bv = (const float*)d_in[4];
    float* out = (float*)d_out;

    // workspace layout (~131 MB total)
    char* p = (char*)d_ws;
    auto take = [&](size_t bytes) { char* r = p; p += (bytes + 255) & ~(size_t)255; return r; };
    unsigned short* xb  = (unsigned short*)take((size_t)B_ * C_ * N_ * 2);    // 64 MiB
    unsigned short* vbw = (unsigned short*)take((size_t)B_ * CV_ * N_ * 2);   // 64 MiB
    unsigned short* Wkb = (unsigned short*)take((size_t)CK_ * C_ * 2);
    unsigned short* Wvb = (unsigned short*)take((size_t)CV_ * C_ * 2);
    float* mpart = (float*)take((size_t)B_ * CK_ * (N_ / 128) * 4);           // 1 MiB
    float* lpart = (float*)take((size_t)B_ * CK_ * (N_ / 128) * 4);
    float* mfull = (float*)take((size_t)B_ * CK_ * 4);
    float* lrec  = (float*)take((size_t)B_ * CK_ * 4);

    k_cvt_w<<<dim3(256), dim3(256), 0, stream>>>(Wk, Wv, Wkb, Wvb);
    k_cvt_x<<<dim3(32768), dim3(256), 0, stream>>>(x, xb);
    k_keys_stats<<<dim3(N_ / 128, 2, B_), dim3(256), SMEM_S, stream>>>(Wkb, xb, bk, mpart, lpart);
    k_values<<<dim3(N_ / 128, 2, B_), dim3(256), SMEM_V, stream>>>(Wvb, xb, bv, vbw);
    k_reduce_stats<<<dim3(8), dim3(256), 0, stream>>>(mpart, lpart, mfull, lrec);
    k_zero<<<dim3((B_ * CK_ * CV_) / 256), dim3(256), 0, stream>>>(out);
    k_context<<<dim3(N_ / 1024, 4, B_), dim3(256), SMEM_C, stream>>>(Wkb, xb, vbw, bk,
                                                                     mfull, lrec, out);
}
